// GATTransformer_69209103007899
// MI455X (gfx1250) — matrix operand order repaired and checked by reading, not compiled, not run on hardware
//
#include <hip/hip_runtime.h>
#include <hip/hip_bf16.h>

typedef __attribute__((ext_vector_type(16))) __bf16 v16bf;
typedef __attribute__((ext_vector_type(8)))  float  v8f;

#define DEV static __device__ __forceinline__

enum { U_ = 2048, O_ = 2048, NI_ = 8000, A_ = 5000,
       TH_ = 768, IH_ = 2048, H_ = 768, HH_ = 256, NALL_ = 17096 };

DEV unsigned short f2bf(float f) {
    unsigned u = __float_as_uint(f);
    unsigned r = 0x7FFFu + ((u >> 16) & 1u);
    return (unsigned short)((u + r) >> 16);
}

DEV float softplusf(float x) {
    if (x > 20.f) return x;
    if (x < -20.f) return expf(x);
    return log1pf(expf(x));
}

DEV unsigned ordenc(float f) {           // order-preserving float->uint map
    unsigned s = __float_as_uint(f);
    return (s & 0x80000000u) ? ~s : (s | 0x80000000u);
}
DEV float orddec(unsigned u) {
    unsigned s = (u & 0x80000000u) ? (u & 0x7FFFFFFFu) : ~u;
    return __uint_as_float(s);
}

// ---------------------------------------------------------------------------
// bf16 WMMA GEMM:  C[MxN] = A[MxK] * Bt[NxK]^T   (both operands row-major,
// K % 32 == 0, N % 64 == 0, A/Bt buffers padded to 64-row multiples so the
// staging loops are completely branch-free wide copies).
// block = 128 threads (4 wave32), block tile 64x64, K step 32, double-
// buffered LDS.
// modes: 0: C = acc + bias[n]
//        1: C = tanh((acc + bias[n]) * scale)
//        2: C = acc
//        3: contrastive reduction: sc = acc*scale;
//           sum += relu(margin - diag[r] + sc) + relu(margin - diag[c] + sc)
//        4: similarity reduction: sum += smat[r,c]*acc - softplus(acc)
// ---------------------------------------------------------------------------
__global__ __launch_bounds__(128)
void k_gemm_wmma(const unsigned short* __restrict__ A,
                 const unsigned short* __restrict__ Bt,
                 float* __restrict__ C, const float* __restrict__ bias,
                 int M, int N, int K, int ldc, int mode, float scale,
                 const float* __restrict__ diag,
                 const float* __restrict__ smat, int smat_ld,
                 float margin, float* __restrict__ accum)
{
    // two ping-pong buffers per operand: 64x32 bf16 = 256 uint4 chunks each
    __shared__ uint4 lds4_a[2 * 256];
    __shared__ uint4 lds4_b[2 * 256];
    __shared__ float redbuf[128];
    unsigned short* lds_a = reinterpret_cast<unsigned short*>(lds4_a);
    unsigned short* lds_b = reinterpret_cast<unsigned short*>(lds4_b);

    const int tid  = threadIdx.x;
    const int lane = tid & 31;
    const int wave = tid >> 5;
    const int wr   = wave >> 1;                      // 2x2 waves of 32x32
    const int wc   = wave & 1;
    const int rowBase = blockIdx.y * 64;
    const int colBase = blockIdx.x * 64;
    const int m16 = lane & 15;
    const int hi  = lane >> 4;

    v8f acc[2][2];
    #pragma unroll
    for (int i = 0; i < 2; ++i)
        #pragma unroll
        for (int j = 0; j < 2; ++j)
            #pragma unroll
            for (int v = 0; v < 8; ++v) acc[i][j][v] = 0.f;

    // chunk geometry: chunk c (0..255) covers tile row (c>>2), 8 bf16 at
    // column (c&3)*8; LDS chunk address is simply chunk index.
    const int c0r = tid >> 2, c0o = (tid & 3) << 3;            // chunk tid
    const int c1r = (tid + 128) >> 2, c1o = ((tid + 128) & 3) << 3;

    uint4 ra0, ra1, rb0, rb1;
    const int nk = K >> 5;

    auto ld_global = [&](int kt) {
        const size_t ko = (size_t)(kt << 5);
        ra0 = *reinterpret_cast<const uint4*>(&A [(size_t)(rowBase + c0r) * K + ko + c0o]);
        ra1 = *reinterpret_cast<const uint4*>(&A [(size_t)(rowBase + c1r) * K + ko + c1o]);
        rb0 = *reinterpret_cast<const uint4*>(&Bt[(size_t)(colBase + c0r) * K + ko + c0o]);
        rb1 = *reinterpret_cast<const uint4*>(&Bt[(size_t)(colBase + c1r) * K + ko + c1o]);
    };
    auto st_lds = [&](int buf) {
        lds4_a[buf * 256 + tid]       = ra0;
        lds4_a[buf * 256 + tid + 128] = ra1;
        lds4_b[buf * 256 + tid]       = rb0;
        lds4_b[buf * 256 + tid + 128] = rb1;
    };

    ld_global(0);
    st_lds(0);
    __syncthreads();

    for (int kt = 0; kt < nk; ++kt) {
        const int cur = kt & 1;
        const bool more = (kt + 1 < nk);
        if (more) ld_global(kt + 1);                 // overlap with WMMA below
        if (kt + 2 < nk) {                           // -> global_prefetch_b8
            const size_t ko = (size_t)((kt + 2) << 5);
            __builtin_prefetch(&A [(size_t)(rowBase + c0r) * K + ko + c0o], 0, 1);
            __builtin_prefetch(&Bt[(size_t)(colBase + c0r) * K + ko + c0o], 0, 1);
        }

        const unsigned short* la = lds_a + cur * 2048;
        const unsigned short* lb = lds_b + cur * 2048;
        union AV { v16bf v; unsigned u[8]; };
        AV afr[2], bfr[2];
        #pragma unroll
        for (int f = 0; f < 2; ++f) {
            // A 16x32 bf16 fragment layout (ISA 7.12.2)
            int r0 = wr * 32 + f * 16 + m16;
            #pragma unroll
            for (int v = 0; v < 8; ++v) {
                int kb = ((v < 4) ? 0 : 16) + (hi << 3) + ((v & 3) << 1);
                afr[f].u[v] = *reinterpret_cast<const unsigned*>(&la[r0 * 32 + kb]);
            }
            // B 32x16 fragment from Bt rows (lane n holds column n of B)
            int n0 = wc * 32 + f * 16 + m16;
            #pragma unroll
            for (int v = 0; v < 8; ++v) {
                int kb = ((v < 4) ? 0 : 16) + (hi << 3) + ((v & 3) << 1);
                bfr[f].u[v] = *reinterpret_cast<const unsigned*>(&lb[n0 * 32 + kb]);
            }
        }
        #pragma unroll
        for (int i = 0; i < 2; ++i)
            #pragma unroll
            for (int j = 0; j < 2; ++j)
                acc[i][j] = __builtin_amdgcn_wmma_f32_16x16x32_bf16(
                    false, afr[i].v, false, bfr[j].v,
                    (short)0, acc[i][j], false, false);

        if (more) {
            __syncthreads();                          // readers of lds[cur] done
            st_lds(1 - cur);
            __syncthreads();                          // new tile visible
        }
    }

    if (mode <= 2) {
        #pragma unroll
        for (int i = 0; i < 2; ++i)
            #pragma unroll
            for (int j = 0; j < 2; ++j)
                #pragma unroll
                for (int v = 0; v < 8; ++v) {
                    int r = rowBase + wr * 32 + i * 16 + v + (hi << 3);
                    int c = colBase + wc * 32 + j * 16 + m16;
                    if (r < M && c < N) {
                        float val = acc[i][j][v];
                        if (mode == 0) val += bias[c];
                        else if (mode == 1) val = tanhf((val + bias[c]) * scale);
                        C[(size_t)r * ldc + c] = val;
                    }
                }
    } else {
        float lsum = 0.f;
        #pragma unroll
        for (int i = 0; i < 2; ++i)
            #pragma unroll
            for (int j = 0; j < 2; ++j)
                #pragma unroll
                for (int v = 0; v < 8; ++v) {
                    int r = rowBase + wr * 32 + i * 16 + v + (hi << 3);
                    int c = colBase + wc * 32 + j * 16 + m16;
                    if (r < M && c < N) {
                        float sc = acc[i][j][v] * scale;
                        if (mode == 3) {
                            float t1 = margin - diag[r] + sc; if (t1 < 0.f) t1 = 0.f;
                            float t2 = margin - diag[c] + sc; if (t2 < 0.f) t2 = 0.f;
                            lsum += t1 + t2;
                        } else {
                            lsum += smat[(size_t)r * smat_ld + c] * sc - softplusf(sc);
                        }
                    }
                }
        __syncthreads();
        redbuf[tid] = lsum;
        __syncthreads();
        for (int s = 64; s > 0; s >>= 1) {
            if (tid < s) redbuf[tid] += redbuf[tid + s];
            __syncthreads();
        }
        if (tid == 0) atomicAdd(accum, redbuf[0]);
    }
}

// ---------------------------------------------------------------------------
// Helper kernels
// ---------------------------------------------------------------------------
__global__ void k_f32_to_bf16(const float* __restrict__ s,
                              unsigned short* __restrict__ d, long n) {
    long i = (long)blockIdx.x * blockDim.x + threadIdx.x;
    long st = (long)gridDim.x * blockDim.x;
    for (; i < n; i += st) d[i] = f2bf(s[i]);
}

__global__ void k_gather_convert(const float* __restrict__ src, int src_ld,
                                 const int* __restrict__ idx, int row_mul,
                                 unsigned short* __restrict__ dst, int dst_ld,
                                 int dst_off, long rows, int cols) {
    long total = rows * (long)cols;
    long st = (long)gridDim.x * blockDim.x;
    for (long i = (long)blockIdx.x * blockDim.x + threadIdx.x; i < total; i += st) {
        long r = i / cols;
        int  c = (int)(i - r * cols);
        long sr = idx ? (long)idx[r] : r * row_mul;
        dst[r * (long)dst_ld + dst_off + c] = f2bf(src[sr * (long)src_ld + c]);
    }
}

__global__ void k_transpose_convert(const float* __restrict__ src, long rows,
                                    int cols, unsigned short* __restrict__ dst) {
    long total = rows * (long)cols;
    long st = (long)gridDim.x * blockDim.x;
    for (long i = (long)blockIdx.x * blockDim.x + threadIdx.x; i < total; i += st) {
        long r = i / cols;
        int  c = (int)(i - r * cols);
        dst[(long)c * rows + r] = f2bf(src[i]);
    }
}

__global__ __launch_bounds__(256)
void k_row_dots(const float* __restrict__ h, int Hc,
                const float* __restrict__ asrc, const float* __restrict__ adst,
                float* __restrict__ ssrc, float* __restrict__ sdst) {
    __shared__ float s1[256], s2[256];
    int row = blockIdx.x, t = threadIdx.x;
    float a = 0.f, b = 0.f;
    for (int c = t; c < Hc; c += 256) {
        float v = h[(size_t)row * Hc + c];
        a += v * asrc[c];
        b += v * adst[c];
    }
    s1[t] = a; s2[t] = b;
    __syncthreads();
    for (int s = 128; s > 0; s >>= 1) {
        if (t < s) { s1[t] += s1[t + s]; s2[t] += s2[t + s]; }
        __syncthreads();
    }
    if (t == 0) { ssrc[row] = s1[0]; sdst[row] = s2[0]; }
}

__global__ void k_edge_max(const int* __restrict__ src, const int* __restrict__ dst,
                           int E, const float* __restrict__ ssrc,
                           const float* __restrict__ sdst,
                           float* __restrict__ e, unsigned* __restrict__ m_u) {
    int i = blockIdx.x * blockDim.x + threadIdx.x;
    if (i >= E) return;
    float v = ssrc[src[i]] + sdst[dst[i]];
    v = (v > 0.f) ? v : 0.2f * v;                    // leaky_relu 0.2
    e[i] = v;
    atomicMax(&m_u[dst[i]], ordenc(v));
}

__global__ void k_edge_exp(const int* __restrict__ dst, int E,
                           const float* __restrict__ e,
                           const unsigned* __restrict__ m_u,
                           float* __restrict__ ex, float* __restrict__ denom) {
    int i = blockIdx.x * blockDim.x + threadIdx.x;
    if (i >= E) return;
    float v = expf(e[i] - orddec(m_u[dst[i]]));
    ex[i] = v;
    atomicAdd(&denom[dst[i]], v);
}

__global__ __launch_bounds__(256)
void k_edge_agg(const int* __restrict__ src, const int* __restrict__ dst,
                const float* __restrict__ ex, const float* __restrict__ h,
                float* __restrict__ agg, int Hc) {
    int i = blockIdx.x;                               // one block per edge
    float w = ex[i];
    int s = src[i], d = dst[i];
    for (int c = threadIdx.x; c < Hc; c += 256)
        atomicAdd(&agg[(size_t)d * Hc + c], w * h[(size_t)s * Hc + c]);
}

__global__ void k_elu_scatter(const float* __restrict__ agg,
                              const float* __restrict__ denom,
                              float* __restrict__ out, long n, int Hc) {
    long total = n * (long)Hc;
    long st = (long)gridDim.x * blockDim.x;
    for (long i = (long)blockIdx.x * blockDim.x + threadIdx.x; i < total; i += st) {
        long r = i / Hc;
        float v = agg[i] / (denom[r] + 1e-16f);
        out[i] = (v > 0.f) ? v : (expf(v) - 1.f);     // elu
    }
}

__global__ __launch_bounds__(256)
void k_diag(const float* __restrict__ im, const float* __restrict__ s, int Hc,
            float inv, float* __restrict__ diag) {
    __shared__ float s1[256];
    int row = blockIdx.x, t = threadIdx.x;
    float a = 0.f;
    for (int c = t; c < Hc; c += 256)
        a += im[(size_t)row * Hc + c] * s[(size_t)row * Hc + c];
    s1[t] = a;
    __syncthreads();
    for (int k = 128; k > 0; k >>= 1) {
        if (t < k) s1[t] += s1[t + k];
        __syncthreads();
    }
    if (t == 0) diag[row] = s1[0] * inv;
}

__global__ __launch_bounds__(256)
void k_normalize(const float* __restrict__ src, int rows, int cols,
                 unsigned short* __restrict__ nrm) {
    __shared__ float s1[256];
    int r = blockIdx.x, t = threadIdx.x;
    float a = 0.f;
    for (int c = t; c < cols; c += 256) {
        float v = src[(size_t)r * cols + c];
        a += v * v;
    }
    s1[t] = a;
    __syncthreads();
    for (int k = 128; k > 0; k >>= 1) {
        if (t < k) s1[t] += s1[t + k];
        __syncthreads();
    }
    float inv = 1.f / fmaxf(sqrtf(s1[0]), 1e-8f);
    for (int c = t; c < cols; c += 256)
        nrm[(size_t)r * cols + c] = f2bf(src[(size_t)r * cols + c] * inv);
}

__global__ __launch_bounds__(256)
void k_bpr(const float* __restrict__ hashv, const int* __restrict__ ui,
           const int* __restrict__ pi, const int* __restrict__ ni,
           int HH, float* __restrict__ accum) {
    __shared__ float s1[256], s2[256];
    int b = blockIdx.x, t = threadIdx.x;
    const float* u = hashv + (size_t)ui[b] * HH;
    const float* p = hashv + (size_t)pi[b] * HH;
    const float* g = hashv + (size_t)ni[b] * HH;
    float a = 0.f, c = 0.f;
    for (int k = t; k < HH; k += 256) {
        float uv = u[k];
        a += uv * p[k];
        c += uv * g[k];
    }
    s1[t] = a; s2[t] = c;
    __syncthreads();
    for (int s = 128; s > 0; s >>= 1) {
        if (t < s) { s1[t] += s1[t + s]; s2[t] += s2[t + s]; }
        __syncthreads();
    }
    if (t == 0) atomicAdd(accum, softplusf(s2[0] - s1[0])); // -log_sigmoid(pos-neg)
}

__global__ void k_final(const float* __restrict__ acc, float* __restrict__ out) {
    float bpr  = acc[0] / 1024.f;
    float vse  = (acc[1] - 2.f * 0.1f * 8000.f) / (7999.f * 8000.f);
    float simo = -acc[2] / (2048.f * 1024.f);
    float simu = -acc[3] / (2048.f * 1024.f);
    out[0] = bpr + vse + 0.1f * simo + simu;
}

// ---------------------------------------------------------------------------
extern "C" void kernel_launch(void* const* d_in, const int* in_sizes, int n_in,
                              void* d_out, int out_size, void* d_ws, size_t ws_size,
                              hipStream_t stream) {
    const float* item_text    = (const float*)d_in[0];
    const float* attr_text    = (const float*)d_in[1];
    const float* image_feat   = (const float*)d_in[2];
    const float* s_oo         = (const float*)d_in[3];
    const float* s_uu         = (const float*)d_in[4];
    const float* user_table   = (const float*)d_in[5];
    const float* outfit_table = (const float*)d_in[6];
    const float* W_user   = (const float*)d_in[7];
    const float* b_user   = (const float*)d_in[8];
    const float* W_outfit = (const float*)d_in[9];
    const float* b_outfit = (const float*)d_in[10];
    const float* W_item   = (const float*)d_in[11];
    const float* b_item   = (const float*)d_in[12];
    const float* W_attr   = (const float*)d_in[13];
    const float* b_attr   = (const float*)d_in[14];
    const float* W_txt    = (const float*)d_in[15];
    const float* b_txt    = (const float*)d_in[16];
    const float* W_hash   = (const float*)d_in[17];
    const float* b_hash   = (const float*)d_in[18];
    const float* W_gat    = (const float*)d_in[19];
    const float* a_src    = (const float*)d_in[20];
    const float* a_dst    = (const float*)d_in[21];
    const int* UO_graph   = (const int*)d_in[22];
    const int* OI_graph   = (const int*)d_in[23];
    const int* IA_graph   = (const int*)d_in[24];
    // d_in[25..27]: UO/OI/IA idx — contiguous aranges, handled via offsets
    const int* user_index    = (const int*)d_in[28];
    const int* pos_index     = (const int*)d_in[29];
    const int* neg_index     = (const int*)d_in[30];
    const int* outfit_emb_ix = (const int*)d_in[31];
    const int* user_emb_ix   = (const int*)d_in[32];

    char* ws = (char*)d_ws;
    size_t off = 0;
    auto alloc = [&](size_t bytes) -> void* {
        size_t o = (off + 255) & ~(size_t)255;
        off = o + bytes;
        return (void*)(ws + o);
    };

    float* node_emb  = (float*)alloc((size_t)NALL_ * H_ * 4);
    float* node_hash = (float*)alloc((size_t)NALL_ * HH_ * 4);
    float* text_emb  = (float*)alloc((size_t)NI_ * H_ * 4);
    float* h_buf     = (float*)alloc((size_t)13000 * H_ * 4);
    float* agg       = (float*)alloc((size_t)13000 * H_ * 4);
    // A buffer padded: max padded use = 17152*768 = 13,172,736 bf16
    unsigned short* A_bf = (unsigned short*)alloc((size_t)13250000 * 2);
    unsigned short* B_bf = (unsigned short*)alloc((size_t)6200000 * 2);  // 8000*768
    unsigned short* nrm_bf = (unsigned short*)alloc((size_t)2048 * 256 * 2);
    unsigned short* w_txt_bf    = (unsigned short*)alloc((size_t)TH_ * H_ * 2);
    unsigned short* w_item_bf   = (unsigned short*)alloc((size_t)2 * H_ * H_ * 2);
    unsigned short* w_attr_bf   = (unsigned short*)alloc((size_t)TH_ * H_ * 2);
    unsigned short* w_user_bf   = (unsigned short*)alloc((size_t)(TH_ + IH_) * H_ * 2);
    unsigned short* w_outfit_bf = (unsigned short*)alloc((size_t)(TH_ + IH_) * H_ * 2);
    unsigned short* w_gat_bf    = (unsigned short*)alloc((size_t)H_ * H_ * 2);
    unsigned short* w_hash_bf   = (unsigned short*)alloc((size_t)H_ * HH_ * 2);
    float*    ssrc  = (float*)alloc(13000 * 4);
    float*    sdst  = (float*)alloc(13000 * 4);
    unsigned* m_u   = (unsigned*)alloc(13000 * 4);
    float*    denom = (float*)alloc(13000 * 4);
    float*    e_buf = (float*)alloc(200000 * 4);
    float*    ex_buf= (float*)alloc(200000 * 4);
    float*    diag  = (float*)alloc(8000 * 4);
    float*    accum = (float*)alloc(16 * 4);

    auto cdiv = [](long a, long b) { return (int)((a + b - 1) / b); };
    auto conv = [&](const float* s, unsigned short* d, long n) {
        int blocks = cdiv(n, 256); if (blocks > 65535) blocks = 65535;
        k_f32_to_bf16<<<blocks, 256, 0, stream>>>(s, d, n);
    };
    // weights: [K][N] f32 -> [N][K] bf16 (Bt operand layout)
    auto convT = [&](const float* s, unsigned short* d, long K, int N) {
        int blocks = cdiv(K * N, 256); if (blocks > 65535) blocks = 65535;
        k_transpose_convert<<<blocks, 256, 0, stream>>>(s, K, N, d);
    };
    auto gemm = [&](const unsigned short* Ab, const unsigned short* Bb, float* C,
                    const float* bias, int M, int N, int K, int ldc, int mode,
                    float scale, const float* dg, const float* sm, int sm_ld,
                    float margin, float* accT) {
        dim3 g((N + 63) / 64, (M + 63) / 64);
        k_gemm_wmma<<<g, 128, 0, stream>>>(Ab, Bb, C, bias, M, N, K, ldc, mode,
                                           scale, dg, sm, sm_ld, margin, accT);
    };

    hipMemsetAsync(accum, 0, 16 * 4, stream);

    // ---- weights -> transposed bf16 ----
    convT(W_txt,    w_txt_bf,    TH_, H_);
    convT(W_item,   w_item_bf,   2 * H_, H_);
    convT(W_attr,   w_attr_bf,   TH_, H_);
    convT(W_user,   w_user_bf,   TH_ + IH_, H_);
    convT(W_outfit, w_outfit_bf, TH_ + IH_, H_);
    convT(W_gat,    w_gat_bf,    H_, H_);
    convT(W_hash,   w_hash_bf,   H_, HH_);

    // ---- text_emb = item_text @ W_txt + b_txt ----
    conv(item_text, A_bf, (long)NI_ * TH_);
    gemm(A_bf, w_txt_bf, text_emb, b_txt, NI_, H_, TH_, H_, 0, 1.f,
         nullptr, nullptr, 0, 0.f, nullptr);

    // ---- item_emb = [image_feat | text_emb] @ W_item + b_item ----
    {
        int blocks = cdiv((long)NI_ * H_, 256);
        k_gather_convert<<<blocks, 256, 0, stream>>>(image_feat, H_, nullptr, 1,
                                                     A_bf, 2 * H_, 0, NI_, H_);
        k_gather_convert<<<blocks, 256, 0, stream>>>(text_emb, H_, nullptr, 1,
                                                     A_bf, 2 * H_, H_, NI_, H_);
    }
    gemm(A_bf, w_item_bf, node_emb + (size_t)(U_ + O_) * H_, b_item,
         NI_, H_, 2 * H_, H_, 0, 1.f, nullptr, nullptr, 0, 0.f, nullptr);

    // ---- attr_emb = attr_text[:,0,:] @ W_attr + b_attr ----
    k_gather_convert<<<cdiv((long)A_ * TH_, 256), 256, 0, stream>>>(
        attr_text, TH_, nullptr, 4, A_bf, TH_, 0, A_, TH_);
    gemm(A_bf, w_attr_bf, node_emb + (size_t)(U_ + O_ + NI_) * H_, b_attr,
         A_, H_, TH_, H_, 0, 1.f, nullptr, nullptr, 0, 0.f, nullptr);

    // ---- outfit_emb / user_emb (gathered table rows) ----
    k_gather_convert<<<cdiv((long)O_ * (TH_ + IH_), 256), 256, 0, stream>>>(
        outfit_table, TH_ + IH_, outfit_emb_ix, 1, A_bf, TH_ + IH_, 0, O_, TH_ + IH_);
    gemm(A_bf, w_outfit_bf, node_emb + (size_t)U_ * H_, b_outfit,
         O_, H_, TH_ + IH_, H_, 0, 1.f, nullptr, nullptr, 0, 0.f, nullptr);

    k_gather_convert<<<cdiv((long)U_ * (TH_ + IH_), 256), 256, 0, stream>>>(
        user_table, TH_ + IH_, user_emb_ix, 1, A_bf, TH_ + IH_, 0, U_, TH_ + IH_);
    gemm(A_bf, w_user_bf, node_emb, b_user,
         U_, H_, TH_ + IH_, H_, 0, 1.f, nullptr, nullptr, 0, 0.f, nullptr);

    // ---- GAT layers: sequence (2,1,0,1,2), contiguous node slices ----
    struct Lay { int base, n, E; const int* src; const int* dst; };
    Lay L[3] = {
        { 0,       U_ + O_,  100000, UO_graph, UO_graph + 100000 },
        { U_,      O_ + NI_, 200000, OI_graph, OI_graph + 200000 },
        { U_ + O_, NI_ + A_,  60000, IA_graph, IA_graph + 60000  },
    };
    const int seq[5] = { 2, 1, 0, 1, 2 };
    for (int s = 0; s < 5; ++s) {
        const Lay& ly = L[seq[s]];
        float* x = node_emb + (size_t)ly.base * H_;
        conv(x, A_bf, (long)ly.n * H_);
        gemm(A_bf, w_gat_bf, h_buf, nullptr, ly.n, H_, H_, H_, 2, 1.f,
             nullptr, nullptr, 0, 0.f, nullptr);
        k_row_dots<<<ly.n, 256, 0, stream>>>(h_buf, H_, a_src, a_dst, ssrc, sdst);
        hipMemsetAsync(m_u,   0, (size_t)ly.n * 4, stream);
        hipMemsetAsync(denom, 0, (size_t)ly.n * 4, stream);
        hipMemsetAsync(agg,   0, (size_t)ly.n * H_ * 4, stream);
        k_edge_max<<<cdiv(ly.E, 256), 256, 0, stream>>>(ly.src, ly.dst, ly.E,
                                                        ssrc, sdst, e_buf, m_u);
        k_edge_exp<<<cdiv(ly.E, 256), 256, 0, stream>>>(ly.dst, ly.E, e_buf,
                                                        m_u, ex_buf, denom);
        k_edge_agg<<<ly.E, 256, 0, stream>>>(ly.src, ly.dst, ex_buf, h_buf, agg, H_);
        k_elu_scatter<<<cdiv((long)ly.n * H_, 256), 256, 0, stream>>>(
            agg, denom, x, ly.n, H_);
    }

    // ---- hash: node_hash = tanh((node_emb @ W_hash + b_hash) * 0.5) ----
    conv(node_emb, A_bf, (long)NALL_ * H_);
    gemm(A_bf, w_hash_bf, node_hash, b_hash, NALL_, HH_, H_, HH_, 1, 0.5f,
         nullptr, nullptr, 0, 0.f, nullptr);

    // ---- BPR loss -> accum[0] ----
    k_bpr<<<1024, 256, 0, stream>>>(node_hash, user_index, pos_index, neg_index,
                                    HH_, accum + 0);

    // ---- contrastive (VSE): fused 8000x8000x768 WMMA reduction -> accum[1]
    //      C = im @ s^T, with Bt convention the B operand is just bf16(text_emb)
    conv(image_feat, A_bf, (long)NI_ * H_);
    conv(text_emb,  B_bf, (long)NI_ * H_);
    k_diag<<<NI_, 256, 0, stream>>>(image_feat, text_emb, H_, 1.f / (float)H_, diag);
    gemm(A_bf, B_bf, nullptr, nullptr, NI_, NI_, H_, 0, 3, 1.f / (float)H_,
         diag, nullptr, 0, 0.1f, accum + 1);

    // ---- similarity losses: fi = nrm @ nrm^T (same operand twice) ----
    k_normalize<<<O_, 256, 0, stream>>>(node_hash + (size_t)U_ * HH_, O_, HH_, nrm_bf);
    gemm(nrm_bf, nrm_bf, nullptr, nullptr, O_, O_, HH_, 0, 4, 1.f,
         nullptr, s_oo, O_, 0.f, accum + 2);

    k_normalize<<<U_, 256, 0, stream>>>(node_hash, U_, HH_, nrm_bf);
    gemm(nrm_bf, nrm_bf, nullptr, nullptr, U_, U_, HH_, 0, 4, 1.f,
         nullptr, s_uu, U_, 0.f, accum + 3);

    // ---- combine ----
    k_final<<<1, 1, 0, stream>>>(accum, (float*)d_out);
    (void)in_sizes; (void)n_in; (void)out_size; (void)ws_size;
}
